// ChaoticLSTM_12738873000430
// MI455X (gfx1250) — matrix operand order repaired and checked by reading, not compiled, not run on hardware
//
#include <hip/hip_runtime.h>
#include <hip/hip_bf16.h>
#include <stdint.h>

// Problem constants (from reference)
#define T_   2048
#define BS_  64
#define IN_  256
#define H_   512
#define G4_  2048            // 4*H
#define KTOT 768             // IN + H combined K
#define NKT  24              // KTOT / 32 k-tiles
#define NWG  16              // persistent workgroups; each owns 32 hidden cols
// per-WG packed weights: 8 n-tiles * NKT k-tiles * 32 lanes * 16 bf16
#define WG_FRAG_ELEMS (8 * NKT * 32 * 16)          // 98304 bf16
#define LDS_BYTES     (8 * NKT * 32 * 32)          // 196608 B

typedef __attribute__((ext_vector_type(16))) __bf16 v16bf;
typedef __attribute__((ext_vector_type(8)))  float  v8f;

union Frag {
    uint4 q[2];
    v16bf v;
};

__device__ __forceinline__ v16bf load_frag(const __hip_bfloat16* p0,
                                           const __hip_bfloat16* p1) {
    Frag f;
    f.q[0] = *(const uint4*)p0;
    f.q[1] = *(const uint4*)p1;
    return f.v;
}

__device__ __forceinline__ v16bf load_frag_lds(const char* p) {
    Frag f;
    f.q[0] = *(const uint4*)p;
    f.q[1] = *(const uint4*)(p + 16);
    return f.v;
}

// A fragment for combined K = [x(256) | h(512)], k-tile kt (compile-time under
// full unroll so the branch folds away).
__device__ __forceinline__ v16bf load_a(int kt, const __hip_bfloat16* xrow,
                                        const __hip_bfloat16* hrow, int koff) {
    if (kt < 8) {
        const __hip_bfloat16* p = xrow + kt * 32 + koff;
        return load_frag(p, p + 16);
    }
    const __hip_bfloat16* p = hrow + (kt - 8) * 32 + koff;
    return load_frag(p, p + 16);
}

__device__ __forceinline__ v16bf load_b(const char* smem, int nt, int kt,
                                        int lane) {
    return load_frag_lds(smem + ((size_t)(nt * NKT + kt) * 32 + lane) * 32);
}

__device__ __forceinline__ float sigf(float x) {
    return 1.0f / (1.0f + __expf(-x));
}
__device__ __forceinline__ float tanha(float x) {
    return 2.0f / (1.0f + __expf(-2.0f * x)) - 1.0f;
}

// ---------------------------------------------------------------------------
// Prologue: zero h ping-pong buffers + sync counter
// ---------------------------------------------------------------------------
__global__ void zero_ws_kernel(unsigned* hbuf_u32, unsigned* cnt) {
    int i = blockIdx.x * 256 + threadIdx.x;
    if (i < (2 * BS_ * H_ * 2) / 4) hbuf_u32[i] = 0u;   // 131072 B
    if (i == 0) *cnt = 0u;
}

// ---------------------------------------------------------------------------
// Prologue: x f32 -> bf16 row-major (A-fragment friendly: rows contiguous)
// ---------------------------------------------------------------------------
__global__ void convert_x_kernel(const float* __restrict__ x,
                                 __hip_bfloat16* __restrict__ xbf, int n4) {
    int i = blockIdx.x * 256 + threadIdx.x;
    if (i >= n4) return;
    int idx = i * 4;
    float4 v = *(const float4*)(x + idx);
    xbf[idx + 0] = __float2bfloat16(v.x);
    xbf[idx + 1] = __float2bfloat16(v.y);
    xbf[idx + 2] = __float2bfloat16(v.z);
    xbf[idx + 3] = __float2bfloat16(v.w);
}

// ---------------------------------------------------------------------------
// Prologue: pack [Wi;Wh] into per-WG WMMA B-fragment order (bf16).
// B 32x16 bf16 layout: lane half h holds K=8h..8h+7 then K=16+8h..16+8h+7
// (col = lane%16); 16 bf16 contiguous per lane.
// Fragment index: ((wg*8 + nt)*NKT + kt)*32 + lane, nt = gate*2 + coltile.
// ---------------------------------------------------------------------------
__global__ void pack_weights_kernel(const float* __restrict__ Wi,
                                    const float* __restrict__ Wh,
                                    __hip_bfloat16* __restrict__ WP) {
    int idx = blockIdx.x * 256 + threadIdx.x;       // one lane-fragment each
    if (idx >= NWG * 8 * NKT * 32) return;
    int lane = idx & 31;
    int kt   = (idx >> 5) % NKT;
    int nt   = (idx / (32 * NKT)) % 8;
    int wg   = idx / (32 * NKT * 8);
    int g    = nt >> 1;
    int ct   = nt & 1;
    int colg = g * H_ + wg * 32 + ct * 16 + (lane & 15);
    int kb   = kt * 32 + ((lane >> 4) * 8);
    __hip_bfloat16* dst = WP + (size_t)idx * 16;
#pragma unroll
    for (int e = 0; e < 16; ++e) {
        int kk = kb + (e < 8 ? e : e + 8);
        float v = (kk < IN_) ? Wi[(size_t)kk * G4_ + colg]
                             : Wh[(size_t)(kk - IN_) * G4_ + colg];
        dst[e] = __float2bfloat16(v);
    }
}

// ---------------------------------------------------------------------------
// Persistent LSTM kernel: 16 WGs x 256 threads (8 waves).
// Wave w: m_tile = w&3 (16 batch rows), ct = w>>2 (16 hidden cols);
// computes i,f,g,o 16x16 tiles for that block; c-state lives in VGPRs.
// Weights resident in LDS; h ping-pongs through global (L2) in bf16.
// Inner GEMM is software-pipelined: fragments for k-tile kt+1 are loaded
// before the four WMMAs of kt (double-buffered, fully unrolled -> no movs).
// ---------------------------------------------------------------------------
__global__ void __launch_bounds__(256, 1)
lstm_persistent_kernel(const __hip_bfloat16* __restrict__ xbf,
                       const __hip_bfloat16* __restrict__ WP,
                       const float* __restrict__ Bias,
                       __hip_bfloat16* hbuf,      // 2 * BS_*H_ bf16
                       unsigned* cnt,
                       float* __restrict__ out) {
    extern __shared__ char smem[];
    const int tid = threadIdx.x;
    const int wg  = blockIdx.x;

    // Stage this WG's weight slice into LDS (192 KB)
    {
        const uint4* src = (const uint4*)(WP + (size_t)wg * WG_FRAG_ELEMS);
        uint4* dst = (uint4*)smem;
        for (int i = tid; i < LDS_BYTES / 16; i += 256) dst[i] = src[i];
    }
    __syncthreads();

    const int lane   = tid & 31;
    const int w      = tid >> 5;
    const int m_tile = w & 3;
    const int ct     = w >> 2;
    const int l15    = lane & 15;
    const int hi     = lane >> 4;
    const int col    = wg * 32 + ct * 16 + l15;   // hidden column this lane owns
    const int koff   = hi * 8;

    float bias[4];
#pragma unroll
    for (int g = 0; g < 4; ++g) bias[g] = Bias[g * H_ + col];

    v8f cst;
#pragma unroll
    for (int r = 0; r < 8; ++r) cst[r] = 0.0f;

    const size_t TBH = (size_t)T_ * BS_ * H_;

    for (int t = 0; t < T_; ++t) {
        const __hip_bfloat16* hread  = hbuf + (size_t)(t & 1) * (BS_ * H_);
        __hip_bfloat16*       hwrite = hbuf + (size_t)((t + 1) & 1) * (BS_ * H_);

        const __hip_bfloat16* xrow =
            xbf + ((size_t)t * BS_ + m_tile * 16 + l15) * IN_;
        const __hip_bfloat16* hrow = hread + (m_tile * 16 + l15) * H_;

        // Zero accumulators: first WMMA per gate folds C as inline constant 0.
        v8f acc[4];
#pragma unroll
        for (int g = 0; g < 4; ++g)
#pragma unroll
            for (int r = 0; r < 8; ++r) acc[g][r] = 0.0f;

        // Software-pipelined GEMM over 24 combined k-tiles.
        v16bf aC = load_a(0, xrow, hrow, koff);
        v16bf bC[4];
#pragma unroll
        for (int g = 0; g < 4; ++g) bC[g] = load_b(smem, g * 2 + ct, 0, lane);

#pragma unroll
        for (int kt = 0; kt < NKT; ++kt) {
            v16bf aN;
            v16bf bN[4];
            if (kt + 1 < NKT) {
                aN = load_a(kt + 1, xrow, hrow, koff);
#pragma unroll
                for (int g = 0; g < 4; ++g)
                    bN[g] = load_b(smem, g * 2 + ct, kt + 1, lane);
            }
#pragma unroll
            for (int g = 0; g < 4; ++g) {
                acc[g] = __builtin_amdgcn_wmma_f32_16x16x32_bf16(
                    false, aC, false, bC[g], (short)0, acc[g], false, false);
            }
            if (kt + 1 < NKT) {
                aC = aN;
#pragma unroll
                for (int g = 0; g < 4; ++g) bC[g] = bN[g];
            }
        }

        // ---- Pointwise LSTM cell update; c stays in registers ----
#pragma unroll
        for (int r = 0; r < 8; ++r) {
            float ig = sigf(acc[0][r] + bias[0]);
            float fg = sigf(acc[1][r] + bias[1]);
            float gg = tanha(acc[2][r] + bias[2]);
            float og = sigf(acc[3][r] + bias[3]);
            float cn = fg * cst[r] + ig * gg;
            cst[r] = cn;
            float hn = og * tanha(cn);
            int row = m_tile * 16 + hi * 8 + r;     // C-layout: VGPR r -> M
            out[((size_t)t * BS_ + row) * H_ + col] = hn;
            hwrite[row * H_ + col] = __float2bfloat16(hn);
            if (t == T_ - 1) {
                out[TBH + (size_t)row * H_ + col] = hn;                    // hT
                out[TBH + (size_t)BS_ * H_ + (size_t)row * H_ + col] = cn; // cT
            }
        }

        // ---- Grid-wide step barrier (ping-pong makes one sync sufficient) ----
        if (t != T_ - 1) {
            __threadfence();
            __syncthreads();
            if (tid == 0) {
                atomicAdd(cnt, 1u);
                const unsigned target = (unsigned)NWG * (unsigned)(t + 1);
                volatile unsigned* vc = cnt;
                while (*vc < target) __builtin_amdgcn_s_sleep(1);
            }
            __syncthreads();
            __threadfence();
        }
    }
}

// ---------------------------------------------------------------------------
// Workspace layout (bytes):
//   [0, 67108864)            xbf   : T*BS*IN bf16
//   [67108864, 70254592)     WP    : packed weights bf16 (3 MB)
//   [70254592, 70385664)     hbuf  : 2 * BS*H bf16 ping-pong
//   [70385664, +4)           cnt   : sync counter
// ---------------------------------------------------------------------------
extern "C" void kernel_launch(void* const* d_in, const int* in_sizes, int n_in,
                              void* d_out, int out_size, void* d_ws, size_t ws_size,
                              hipStream_t stream) {
    (void)in_sizes; (void)n_in; (void)out_size; (void)ws_size;
    const float* x  = (const float*)d_in[0];
    const float* Wi = (const float*)d_in[1];
    const float* Wh = (const float*)d_in[2];
    const float* Bv = (const float*)d_in[3];
    float* out = (float*)d_out;

    char* ws = (char*)d_ws;
    __hip_bfloat16* xbf  = (__hip_bfloat16*)(ws);
    __hip_bfloat16* WP   = (__hip_bfloat16*)(ws + 67108864);
    __hip_bfloat16* hbuf = (__hip_bfloat16*)(ws + 70254592);
    unsigned*       cnt  = (unsigned*)(ws + 70385664);

    zero_ws_kernel<<<128, 256, 0, stream>>>((unsigned*)hbuf, cnt);

    int n4 = (T_ * BS_ * IN_) / 4;
    convert_x_kernel<<<(n4 + 255) / 256, 256, 0, stream>>>(x, xbf, n4);

    int nfrag = NWG * 8 * NKT * 32;
    pack_weights_kernel<<<(nfrag + 255) / 256, 256, 0, stream>>>(Wi, Wh, WP);

    lstm_persistent_kernel<<<NWG, 256, LDS_BYTES, stream>>>(
        xbf, WP, Bv, hbuf, cnt, out);
}
